// EncoderLayer_24077586661984
// MI455X (gfx1250) — compile-verified, matrix operand order checked by reading, not run on hardware
//
#include <hip/hip_runtime.h>
#include <math.h>

// ---------------------------------------------------------------------------
// Encoder layer for MI455X (gfx1250), wave32 + WMMA bf16 (f32 accumulate).
//   B=4, S=2048, D=768, H=12, DH=64, DFF=3072, M = B*S = 8192.
// GEMMs + both attention contractions on v_wmma_f32_16x16x32_bf16.
// GEMM tile staging uses the Tensor Data Mover (tensor_load_to_lds) with
// TENSORcnt double-buffering; flash attention keeps scores in registers.
// ---------------------------------------------------------------------------

typedef __attribute__((ext_vector_type(16))) __bf16 bf16x16;
typedef __attribute__((ext_vector_type(8)))  __bf16 bf16x8;
typedef __attribute__((ext_vector_type(8)))  float  f32x8;
typedef unsigned int u32x4 __attribute__((ext_vector_type(4)));
typedef int          i32x4 __attribute__((ext_vector_type(4)));
typedef int          i32x8 __attribute__((ext_vector_type(8)));

#define B_    4
#define S_    2048
#define D_    768
#define H_    12
#define DH_   64
#define DFF_  3072
#define MROWS (B_ * S_)   // 8192

#if defined(__has_builtin)
#if __has_builtin(__builtin_amdgcn_tensor_load_to_lds) && \
    __has_builtin(__builtin_amdgcn_s_wait_tensorcnt)
#define HAVE_TDM 1
#endif
#endif
#ifndef HAVE_TDM
#define HAVE_TDM 0
#endif

// D = A(16x32 bf16) x B(32x16 bf16) + C(16x16 f32)
__device__ __forceinline__ f32x8 wmma_bf16(bf16x16 a, bf16x16 b, f32x8 c) {
  return __builtin_amdgcn_wmma_f32_16x16x32_bf16(
      /*neg_a=*/false, a, /*neg_b=*/false, b,
      /*c_mod=*/(short)0, c, /*reuse_a=*/false, /*reuse_b=*/false);
}

// 16-bit A/B fragment per ISA layout: element e of the v16 holds
//   K = (e < 8 ? 0 : 16) + 8*(lane/16) + (e % 8)
// i.e. two contiguous 8-element (16-byte) chunks at +hi*8 and +16+hi*8.
__device__ __forceinline__ bf16x16 frag_ld(const __bf16* rowPtr, int hi) {
  bf16x8 lo  = *(const bf16x8*)(rowPtr + hi * 8);
  bf16x8 hiv = *(const bf16x8*)(rowPtr + 16 + hi * 8);
  bf16x16 r;
#pragma unroll
  for (int i = 0; i < 8; ++i) { r[i] = lo[i]; r[i + 8] = hiv[i]; }
  return r;
}

#if HAVE_TDM
typedef __attribute__((address_space(3))) __bf16 lds_bf16_t;

// Issue one 2D-tile TDM load: rows x cols (bf16) from a row-major tensor with
// row stride `rowStride` (elements) into contiguous LDS [rows][cols].
// Descriptor bit layout per CDNA5 ISA §8.3 (group0) / §8.4 (group1).
__device__ __forceinline__ void tdm_load_2d(const __bf16* gsrc, __bf16* ldst,
                                            int rows, int cols, int rowStride,
                                            int tensorRows) {
  const unsigned long long ga = (unsigned long long)gsrc;
  const unsigned lds =
      (unsigned)(unsigned long long)(lds_bf16_t*)ldst;  // LDS byte address
  u32x4 g0 = {
      1u,                                                  // count=1 (valid)
      lds,                                                 // lds_addr
      (unsigned)(ga & 0xFFFFFFFFull),                      // global_addr[31:0]
      (unsigned)((ga >> 32) & 0x1FFFFFFull) | 0x80000000u  // addr[56:32]|type=2
  };
  i32x8 g1;
  g1[0] = 0x00010000;                                  // data_size=2B, no mask
  g1[1] = (int)(((unsigned)rowStride & 0xFFFFu) << 16);   // tensor_dim0[15:0]
  g1[2] = (int)((((unsigned)rowStride >> 16) & 0xFFFFu) |
                (((unsigned)tensorRows & 0xFFFFu) << 16)); // dim0 hi | dim1 lo
  g1[3] = (int)((((unsigned)tensorRows >> 16) & 0xFFFFu) |
                ((unsigned)cols << 16));               // dim1 hi | tile_dim0
  g1[4] = rows & 0xFFFF;                               // tile_dim1 (tile_dim2=0)
  g1[5] = rowStride;                                   // tensor_dim0_stride lo
  g1[6] = 0;                                           // stride hi, dim1_stride
  g1[7] = 0;
  i32x4 g2 = (i32x4)0;                                 // dims 2/3 unused
  i32x4 g3 = (i32x4)0;
#if __clang_major__ >= 23
  i32x8 g4 = (i32x8)0;
  __builtin_amdgcn_tensor_load_to_lds(g0, g1, g2, g3, g4, 0);
#else
  __builtin_amdgcn_tensor_load_to_lds(g0, g1, g2, g3, 0);
#endif
}
#endif  // HAVE_TDM

// ---------------------------------------------------------------------------
// f32 -> bf16 conversion (weights + activations)
// ---------------------------------------------------------------------------
__global__ __launch_bounds__(256) void cvt_f32_bf16(const float* __restrict__ in,
                                                    __bf16* __restrict__ out, int n) {
  for (int i = blockIdx.x * 256 + threadIdx.x; i < n; i += gridDim.x * 256)
    out[i] = (__bf16)in[i];
}

// ---------------------------------------------------------------------------
// Generic GEMM:  out[M,N] = A[M,K](bf16) * W[N,K]^T(bf16) + bias (+residual)(+relu)
// Block: 256 threads = 8 waves, tile 128x128; wave tile 32x64 (2x4 wmma tiles).
// Tiles staged by the Tensor Data Mover, double-buffered on TENSORcnt.
// ---------------------------------------------------------------------------
__global__ __launch_bounds__(256) void gemm_bf16_nt(
    const __bf16* __restrict__ A, const __bf16* __restrict__ W,
    const float* __restrict__ bias, const float* __restrict__ residual,
    float* __restrict__ outF, __bf16* __restrict__ outB,
    int M, int N, int K, int doRelu)
{
  __shared__ __align__(16) __bf16 As[2][128 * 32];
  __shared__ __align__(16) __bf16 Bs[2][128 * 32];

  const int tid   = threadIdx.x;
  const int wave  = tid >> 5;
  const int lane  = tid & 31;
  const int ln16  = lane & 15;
  const int hi    = lane >> 4;
  const int waveM = wave >> 1;        // 0..3 : 32 rows each
  const int waveN = wave & 1;         // 0..1 : 64 cols each
  const int blockM = blockIdx.y * 128;
  const int blockN = blockIdx.x * 128;
  const int T = K >> 5;               // number of 32-wide K tiles

  const f32x8 z = {0.f, 0.f, 0.f, 0.f, 0.f, 0.f, 0.f, 0.f};
  f32x8 acc[2][4];
#pragma unroll
  for (int mt = 0; mt < 2; ++mt)
#pragma unroll
    for (int nt = 0; nt < 4; ++nt) acc[mt][nt] = z;

  const __bf16* aBase = A + (size_t)blockM * K;
  const __bf16* wBase = W + (size_t)blockN * K;

#if HAVE_TDM
  // ---- TDM double-buffered pipeline (wave 0 drives the DMA engine) ----
  if (wave == 0) {
    tdm_load_2d(aBase, &As[0][0], 128, 32, K, M);
    tdm_load_2d(wBase, &Bs[0][0], 128, 32, K, N);
  }
  for (int t = 0; t < T; ++t) {
    const int buf = t & 1;
    if (wave == 0) {
      if (t + 1 < T) {
        tdm_load_2d(aBase + (t + 1) * 32, &As[buf ^ 1][0], 128, 32, K, M);
        tdm_load_2d(wBase + (t + 1) * 32, &Bs[buf ^ 1][0], 128, 32, K, N);
        __builtin_amdgcn_s_wait_tensorcnt(2);  // tile t's two DMAs done
      } else {
        __builtin_amdgcn_s_wait_tensorcnt(0);
      }
    }
    __syncthreads();  // tile t visible to all waves

    bf16x16 aF[2], bF[4];
#pragma unroll
    for (int mt = 0; mt < 2; ++mt)
      aF[mt] = frag_ld(&As[buf][(waveM * 32 + mt * 16 + ln16) * 32], hi);
#pragma unroll
    for (int nt = 0; nt < 4; ++nt)
      bF[nt] = frag_ld(&Bs[buf][(waveN * 64 + nt * 16 + ln16) * 32], hi);
#pragma unroll
    for (int mt = 0; mt < 2; ++mt)
#pragma unroll
      for (int nt = 0; nt < 4; ++nt)
        acc[mt][nt] = wmma_bf16(aF[mt], bF[nt], acc[mt][nt]);

    __syncthreads();  // done with buf before it is re-targeted at t+1
  }
#else
  // ---- fallback: register double-buffering (batched loads, no TDM) ----
  const int ldRow = tid >> 1;
  const int ldCol = (tid & 1) << 4;
  const __bf16* aSrc = aBase + (size_t)ldRow * K + ldCol;
  const __bf16* wSrc = wBase + (size_t)ldRow * K + ldCol;
  bf16x8 ra0 = *(const bf16x8*)(aSrc);
  bf16x8 ra1 = *(const bf16x8*)(aSrc + 8);
  bf16x8 rb0 = *(const bf16x8*)(wSrc);
  bf16x8 rb1 = *(const bf16x8*)(wSrc + 8);
  for (int t = 0; t < T; ++t) {
    __syncthreads();
    *(bf16x8*)&As[0][ldRow * 32 + ldCol]     = ra0;
    *(bf16x8*)&As[0][ldRow * 32 + ldCol + 8] = ra1;
    *(bf16x8*)&Bs[0][ldRow * 32 + ldCol]     = rb0;
    *(bf16x8*)&Bs[0][ldRow * 32 + ldCol + 8] = rb1;
    __syncthreads();
    if (t + 1 < T) {  // overlap next-tile global loads with WMMA below
      ra0 = *(const bf16x8*)(aSrc + (t + 1) * 32);
      ra1 = *(const bf16x8*)(aSrc + (t + 1) * 32 + 8);
      rb0 = *(const bf16x8*)(wSrc + (t + 1) * 32);
      rb1 = *(const bf16x8*)(wSrc + (t + 1) * 32 + 8);
      __builtin_prefetch(aSrc + (t + 2) * 32, 0, 3);
      __builtin_prefetch(wSrc + (t + 2) * 32, 0, 3);
    }
    bf16x16 aF[2], bF[4];
#pragma unroll
    for (int mt = 0; mt < 2; ++mt)
      aF[mt] = frag_ld(&As[0][(waveM * 32 + mt * 16 + ln16) * 32], hi);
#pragma unroll
    for (int nt = 0; nt < 4; ++nt)
      bF[nt] = frag_ld(&Bs[0][(waveN * 64 + nt * 16 + ln16) * 32], hi);
#pragma unroll
    for (int mt = 0; mt < 2; ++mt)
#pragma unroll
      for (int nt = 0; nt < 4; ++nt)
        acc[mt][nt] = wmma_bf16(aF[mt], bF[nt], acc[mt][nt]);
  }
#endif

  // Epilogue. C layout: lane%16 = N, VGPR v -> M = v + 8*(lane/16).
#pragma unroll
  for (int mt = 0; mt < 2; ++mt) {
#pragma unroll
    for (int nt = 0; nt < 4; ++nt) {
      const int col = blockN + waveN * 64 + nt * 16 + ln16;
      const float bc = bias ? bias[col] : 0.f;
#pragma unroll
      for (int v = 0; v < 8; ++v) {
        const int row = blockM + waveM * 32 + mt * 16 + v + 8 * hi;
        float x = acc[mt][nt][v] + bc;
        if (residual) x += residual[(size_t)row * N + col];
        if (doRelu)   x = fmaxf(x, 0.f);
        if (outF) outF[(size_t)row * N + col] = x;
        if (outB) outB[(size_t)row * N + col] = (__bf16)x;
      }
    }
  }
}

// ---------------------------------------------------------------------------
// Flash attention: block = 128 threads (4 waves), one (b,h) x 64 query rows.
// Each wave owns 16 query rows; kv tiles of 64 keys.
// ---------------------------------------------------------------------------
__global__ __launch_bounds__(128) void flash_attn_bf16(
    const __bf16* __restrict__ Qp, const __bf16* __restrict__ Kp,
    const __bf16* __restrict__ Vp, const unsigned char* __restrict__ maskP,
    __bf16* __restrict__ Op)
{
  const int bh   = blockIdx.y;         // 0..47
  const int b    = bh / H_;
  const int h    = bh % H_;
  const int tid  = threadIdx.x;
  const int wave = tid >> 5;
  const int lane = tid & 31;
  const int ln16 = lane & 15;
  const int hi   = lane >> 4;
  const int q0   = blockIdx.x * 64 + wave * 16;   // first query row of this wave

  __shared__ __align__(16) __bf16 Vt[64 * 64];        // V^T tile: [dh][kv]
  __shared__ __align__(16) __bf16 Pl[4 * 16 * 64];    // per-wave P tile: [q][kv]

  // Q fragments (held across the whole kv loop), pre-scaled by 1/sqrt(DH).
  bf16x16 qF[2];
#pragma unroll
  for (int ks = 0; ks < 2; ++ks) {
    const __bf16* p = Qp + (size_t)(b * S_ + q0 + ln16) * D_ + h * DH_ + ks * 32;
    bf16x16 t = frag_ld(p, hi);
#pragma unroll
    for (int i = 0; i < 16; ++i) t[i] = (__bf16)((float)t[i] * 0.125f);
    qF[ks] = t;
  }

  const f32x8 z = {0.f, 0.f, 0.f, 0.f, 0.f, 0.f, 0.f, 0.f};
  f32x8 accO[4];
#pragma unroll
  for (int nt = 0; nt < 4; ++nt) accO[nt] = z;
  float mrun[8], lrun[8];
#pragma unroll
  for (int v = 0; v < 8; ++v) { mrun[v] = -3.0e38f; lrun[v] = 0.f; }

  for (int kv0 = 0; kv0 < S_; kv0 += 64) {
    // Stage V tile transposed ([dh][kv]): 16B row chunks, 8 ds_store_b16 each.
    __syncthreads();
    for (int i = tid; i < 512; i += 128) {
      const int kk = i >> 3;            // kv row within tile
      const int c8 = (i & 7) * 8;       // dh chunk start
      bf16x8 vrow =
          *(const bf16x8*)(Vp + (size_t)(b * S_ + kv0 + kk) * D_ + h * DH_ + c8);
#pragma unroll
      for (int j = 0; j < 8; ++j) Vt[(c8 + j) * 64 + kk] = vrow[j];
    }
    __syncthreads();

    // ---- scores S = (Q/sqrt(dh)) · K^T : 16q x 64k per wave ----
    f32x8 s[4];
#pragma unroll
    for (int nt = 0; nt < 4; ++nt) {
      f32x8 c = z;
#pragma unroll
      for (int ks = 0; ks < 2; ++ks) {
        const __bf16* kp =
            Kp + (size_t)(b * S_ + kv0 + nt * 16 + ln16) * D_ + h * DH_ + ks * 32;
        c = wmma_bf16(qF[ks], frag_ld(kp, hi), c);
      }
      s[nt] = c;
    }

    // ---- attention mask (bool [B,1,S,S]) ----
#pragma unroll
    for (int nt = 0; nt < 4; ++nt) {
      const int key = kv0 + nt * 16 + ln16;
#pragma unroll
      for (int v = 0; v < 8; ++v) {
        const int q = q0 + v + 8 * hi;
        if (!maskP[((size_t)b * S_ + q) * S_ + key]) s[nt][v] = -3.0e38f;
      }
    }

    // ---- online softmax (rows live across 16-lane groups) ----
    float mnew[8], scl[8];
#pragma unroll
    for (int v = 0; v < 8; ++v) {
      float m = fmaxf(fmaxf(s[0][v], s[1][v]), fmaxf(s[2][v], s[3][v]));
#pragma unroll
      for (int off = 1; off < 16; off <<= 1) m = fmaxf(m, __shfl_xor(m, off, 32));
      const float mn = fmaxf(mrun[v], m);
      scl[v]  = __expf(mrun[v] - mn);
      mrun[v] = mn;
      mnew[v] = mn;
    }
#pragma unroll
    for (int v = 0; v < 8; ++v) {
      float rs = 0.f;
#pragma unroll
      for (int nt = 0; nt < 4; ++nt) {
        const float p = __expf(s[nt][v] - mnew[v]);
        s[nt][v] = p;
        rs += p;
      }
#pragma unroll
      for (int off = 1; off < 16; off <<= 1) rs += __shfl_xor(rs, off, 32);
      lrun[v] = lrun[v] * scl[v] + rs;
    }
#pragma unroll
    for (int nt = 0; nt < 4; ++nt)
#pragma unroll
      for (int v = 0; v < 8; ++v) accO[nt][v] *= scl[v];

    // ---- re-layout P (C layout -> A layout) through wave-private LDS ----
#pragma unroll
    for (int nt = 0; nt < 4; ++nt)
#pragma unroll
      for (int v = 0; v < 8; ++v)
        Pl[wave * 1024 + (v + 8 * hi) * 64 + nt * 16 + ln16] = (__bf16)s[nt][v];
    // Same-wave LDS RAW: LDS is in-order per wave; the explicit wait keeps the
    // compiler from reordering and makes the dependency explicit.
    asm volatile("s_wait_dscnt 0x0" ::: "memory");

    bf16x16 pF[2];
#pragma unroll
    for (int ks = 0; ks < 2; ++ks)
      pF[ks] = frag_ld(&Pl[wave * 1024 + ln16 * 64 + ks * 32], hi);

    // ---- accO += P · V ----
#pragma unroll
    for (int nt = 0; nt < 4; ++nt)
#pragma unroll
      for (int ks = 0; ks < 2; ++ks)
        accO[nt] = wmma_bf16(pF[ks], frag_ld(&Vt[(nt * 16 + ln16) * 64 + ks * 32], hi),
                             accO[nt]);
  }

  // ---- finalize: O = acc / l, write bf16 [B*S, D] at head offset ----
#pragma unroll
  for (int nt = 0; nt < 4; ++nt) {
#pragma unroll
    for (int v = 0; v < 8; ++v) {
      const int q = q0 + v + 8 * hi;
      const float o = accO[nt][v] / fmaxf(lrun[v], 1e-30f);
      Op[(size_t)(b * S_ + q) * D_ + h * DH_ + nt * 16 + ln16] = (__bf16)o;
    }
  }
}

// ---------------------------------------------------------------------------
// LayerNorm (faithful: ddof=1 std, eps added to std). One block per row.
// ---------------------------------------------------------------------------
__global__ __launch_bounds__(256) void layernorm_row(
    const float* __restrict__ x, const float* __restrict__ alpha,
    const float* __restrict__ beta, float* __restrict__ outF,
    __bf16* __restrict__ outB)
{
  const int row = blockIdx.x;
  const int tid = threadIdx.x;
  const float* xr = x + (size_t)row * D_;

  float s = 0.f, ss = 0.f;
  for (int c = tid; c < D_; c += 256) { const float v = xr[c]; s += v; ss += v * v; }

  __shared__ float r1[256], r2[256];
  r1[tid] = s; r2[tid] = ss;
  __syncthreads();
  for (int off = 128; off > 0; off >>= 1) {
    if (tid < off) { r1[tid] += r1[tid + off]; r2[tid] += r2[tid + off]; }
    __syncthreads();
  }
  const float mean = r1[0] * (1.f / (float)D_);
  const float var  = fmaxf(r2[0] - (float)D_ * mean * mean, 0.f) / (float)(D_ - 1);
  const float inv  = 1.f / (sqrtf(var) + 1e-6f);

  for (int c = tid; c < D_; c += 256) {
    const float v = alpha[c] * (xr[c] - mean) * inv + beta[c];
    if (outF) outF[(size_t)row * D_ + c] = v;
    if (outB) outB[(size_t)row * D_ + c] = (__bf16)v;
  }
}

// ---------------------------------------------------------------------------
// Host-side orchestration
// ---------------------------------------------------------------------------
extern "C" void kernel_launch(void* const* d_in, const int* in_sizes, int n_in,
                              void* d_out, int out_size, void* d_ws, size_t ws_size,
                              hipStream_t stream) {
  (void)in_sizes; (void)n_in; (void)out_size; (void)ws_size;

  const float*         src   = (const float*)d_in[0];
  const unsigned char* amask = (const unsigned char*)d_in[1];
  const float* Wq = (const float*)d_in[2];   const float* bq = (const float*)d_in[3];
  const float* Wk = (const float*)d_in[4];   const float* bk = (const float*)d_in[5];
  const float* Wv = (const float*)d_in[6];   const float* bv = (const float*)d_in[7];
  const float* Wo = (const float*)d_in[8];   const float* bo = (const float*)d_in[9];
  const float* W1 = (const float*)d_in[10];  const float* b1 = (const float*)d_in[11];
  const float* W2 = (const float*)d_in[12];  const float* b2 = (const float*)d_in[13];
  const float* ln0a = (const float*)d_in[14]; const float* ln0b = (const float*)d_in[15];
  const float* ln1a = (const float*)d_in[16]; const float* ln1b = (const float*)d_in[17];
  float* out = (float*)d_out;

  // ---- workspace carve-up ----
  char* w = (char*)d_ws;
  auto take = [&](size_t bytes) -> void* {
    void* p = (void*)w;
    w += (bytes + 255) & ~(size_t)255;
    return p;
  };
  const size_t SX = (size_t)MROWS * D_;          // 8192*768
  __bf16* bfX  = (__bf16*)take(SX * 2);          // src in bf16
  __bf16* bfQ  = (__bf16*)take(SX * 2);
  __bf16* bfK  = (__bf16*)take(SX * 2);
  __bf16* bfV  = (__bf16*)take(SX * 2);
  __bf16* bfA  = (__bf16*)take(SX * 2);          // attention output
  __bf16* bfX1 = (__bf16*)take(SX * 2);          // LN0 output
  __bf16* bfHH = (__bf16*)take((size_t)MROWS * DFF_ * 2);
  float*  fT0  = (float*)take(SX * 4);           // Wo out + residual (LN0 in)
  float*  fT1  = (float*)take(SX * 4);           // FFN2 out (LN1 in)
  __bf16* bWq  = (__bf16*)take((size_t)D_ * D_ * 2);
  __bf16* bWk  = (__bf16*)take((size_t)D_ * D_ * 2);
  __bf16* bWv  = (__bf16*)take((size_t)D_ * D_ * 2);
  __bf16* bWo  = (__bf16*)take((size_t)D_ * D_ * 2);
  __bf16* bW1  = (__bf16*)take((size_t)DFF_ * D_ * 2);
  __bf16* bW2  = (__bf16*)take((size_t)D_ * DFF_ * 2);

  // ---- 1) f32 -> bf16 conversions ----
  auto cvt = [&](const float* in, __bf16* o, size_t n) {
    int blocks = (int)((n + 255) / 256); if (blocks > 4096) blocks = 4096;
    cvt_f32_bf16<<<blocks, 256, 0, stream>>>(in, o, (int)n);
  };
  cvt(src, bfX, SX);
  cvt(Wq, bWq, (size_t)D_ * D_);
  cvt(Wk, bWk, (size_t)D_ * D_);
  cvt(Wv, bWv, (size_t)D_ * D_);
  cvt(Wo, bWo, (size_t)D_ * D_);
  cvt(W1, bW1, (size_t)DFF_ * D_);
  cvt(W2, bW2, (size_t)D_ * DFF_);

  const dim3 blk(256);
  const dim3 gD (D_   / 128, MROWS / 128);   // (6, 64)
  const dim3 gF1(DFF_ / 128, MROWS / 128);   // (24, 64)

  // ---- 2) Q/K/V projections (bf16 out) ----
  gemm_bf16_nt<<<gD, blk, 0, stream>>>(bfX, bWq, bq, nullptr, nullptr, bfQ,
                                       MROWS, D_, D_, 0);
  gemm_bf16_nt<<<gD, blk, 0, stream>>>(bfX, bWk, bk, nullptr, nullptr, bfK,
                                       MROWS, D_, D_, 0);
  gemm_bf16_nt<<<gD, blk, 0, stream>>>(bfX, bWv, bv, nullptr, nullptr, bfV,
                                       MROWS, D_, D_, 0);

  // ---- 3) flash attention ----
  flash_attn_bf16<<<dim3(S_ / 64, B_ * H_), dim3(128), 0, stream>>>(
      bfQ, bfK, bfV, amask, bfA);

  // ---- 4) output projection + residual (f32 out) ----
  gemm_bf16_nt<<<gD, blk, 0, stream>>>(bfA, bWo, bo, src, fT0, nullptr,
                                       MROWS, D_, D_, 0);

  // ---- 5) LN0 -> bf16 ----
  layernorm_row<<<MROWS, blk, 0, stream>>>(fT0, ln0a, ln0b, nullptr, bfX1);

  // ---- 6) FFN1 (+ReLU, bf16 out) ----
  gemm_bf16_nt<<<gF1, blk, 0, stream>>>(bfX1, bW1, b1, nullptr, nullptr, bfHH,
                                        MROWS, DFF_, D_, 1);

  // ---- 7) FFN2 (f32 out) ----
  gemm_bf16_nt<<<gD, blk, 0, stream>>>(bfHH, bW2, b2, nullptr, fT1, nullptr,
                                       MROWS, D_, DFF_, 0);

  // ---- 8) LN1 -> d_out (f32) ----
  layernorm_row<<<MROWS, blk, 0, stream>>>(fT1, ln1a, ln1b, out, nullptr);
}
